// Attention_50105088475440
// MI455X (gfx1250) — matrix operand order repaired and checked by reading, not compiled, not run on hardware
//
#include <hip/hip_runtime.h>

typedef __attribute__((ext_vector_type(16))) __bf16 v16bf;
typedef __attribute__((ext_vector_type(8)))  float  v8f;

#define SEQ 2048
#define DIM 4096
#define NH  32
#define NKV 8
#define HD  128

__device__ __forceinline__ unsigned short f2bf(float f) {
  unsigned int u = __float_as_uint(f);
  u += 0x7FFFu + ((u >> 16) & 1u);          // round-to-nearest-even
  return (unsigned short)(u >> 16);
}

union Frag {
  v16bf v;
  uint4 q[2];
  unsigned short u[16];
};

// ---- CDNA5 async global->LDS copy (ASYNCcnt path), 16B per lane ----
__device__ __forceinline__ void async_copy_b128(unsigned int lds_off,
                                                const void* gptr) {
  asm volatile("global_load_async_to_lds_b128 %0, %1, off"
               :: "v"(lds_off), "v"(gptr) : "memory");
}
__device__ __forceinline__ void wait_async0() {
  asm volatile("s_wait_asynccnt 0x0" ::: "memory");
}
__device__ __forceinline__ void wait_async4() {
  asm volatile("s_wait_asynccnt 0x4" ::: "memory");
}
__device__ __forceinline__ void wait_async8() {
  asm volatile("s_wait_asynccnt 0x8" ::: "memory");
}
__device__ __forceinline__ unsigned int lds_off(const void* p) {
  return (unsigned int)(unsigned long long)p;
}

// ---------------- elementwise f32 -> bf16 ----------------
__global__ void cvt_bf16_kernel(const float* __restrict__ src,
                                unsigned short* __restrict__ dst, int n) {
  int i = blockIdx.x * blockDim.x + threadIdx.x;
  int stride = gridDim.x * blockDim.x;
  for (; i < n; i += stride) dst[i] = f2bf(src[i]);
}

// ---------------- tiled transpose + f32 -> bf16 ----------------
// src: R x C (f32, row-major) -> dst: C x R (bf16, row-major)
__global__ __launch_bounds__(256) void transpose_cvt_kernel(
    const float* __restrict__ src, unsigned short* __restrict__ dst,
    int R, int C) {
  __shared__ float tile[32][33];
  int bx = blockIdx.x * 32;                 // column base (C dim)
  int by = blockIdx.y * 32;                 // row base (R dim)
  int tx = threadIdx.x & 31, ty = threadIdx.x >> 5;   // ty: 0..7
#pragma unroll
  for (int j = 0; j < 32; j += 8)
    tile[ty + j][tx] = src[(size_t)(by + ty + j) * C + bx + tx];
  __syncthreads();
#pragma unroll
  for (int j = 0; j < 32; j += 8)
    dst[(size_t)(bx + ty + j) * R + by + tx] = f2bf(tile[tx][ty + j]);
}

// ---------------- RoPE (interleaved pairs) + bf16 convert ----------------
__global__ void rope_kernel(const float* __restrict__ src,
                            unsigned short* __restrict__ dst,
                            const float* __restrict__ cosb,
                            const float* __restrict__ sinb,
                            int nheads, int total) {
  int i = blockIdx.x * blockDim.x + threadIdx.x;
  if (i >= total) return;
  int j = i & 63;                 // HD/2 = 64
  int s = i / (64 * nheads);
  float a = src[2 * i], b = src[2 * i + 1];
  float c = cosb[s * 64 + j], sn = sinb[s * 64 + j];
  dst[2 * i]     = f2bf(a * c - b * sn);
  dst[2 * i + 1] = f2bf(a * sn + b * c);
}

// ---------------- bf16 TN GEMM via WMMA, async double-buffered ----------------
// C[M,N] = A[M,K] * Bt[N,K]^T ; A row-major MxK, Bt row-major NxK (pre-transposed B)
__global__ __launch_bounds__(256) void gemm_bf16_tn_kernel(
    const unsigned short* __restrict__ A, const unsigned short* __restrict__ Bt,
    float* __restrict__ C, int M, int N, int K) {
  __shared__ alignas(16) unsigned short As[2][128 * 32];   // [m][k]
  __shared__ alignas(16) unsigned short Bs[2][128 * 32];   // [n][k]
  int tid = threadIdx.x;
  int wid = tid >> 5;
  int lane = tid & 31, ln = lane & 15, hi = lane >> 4;
  int m0 = blockIdx.y * 128, n0 = blockIdx.x * 128;

  v8f acc[8];
#pragma unroll
  for (int t = 0; t < 8; ++t)
#pragma unroll
    for (int r = 0; r < 8; ++r) acc[t][r] = 0.0f;

  // per-thread share of one tile pair = 4 async b128 copies
  auto stage = [&](int k0, int buf) {
#pragma unroll
    for (int i = 0; i < 2; ++i) {
      int idx = tid + 256 * i;              // 0..511
      int r = idx >> 2, c4 = idx & 3;       // row 0..127, 8-col chunk 0..3
      async_copy_b128(lds_off(&As[buf][r * 32 + c4 * 8]),
                      &A[(size_t)(m0 + r) * K + k0 + c4 * 8]);
      async_copy_b128(lds_off(&Bs[buf][r * 32 + c4 * 8]),
                      &Bt[(size_t)(n0 + r) * K + k0 + c4 * 8]);
    }
  };

  auto compute = [&](int cur) {
    Frag af;
    int m = wid * 16 + ln;
    af.q[0] = *(const uint4*)&As[cur][m * 32 + 8 * hi];       // k = 8*hi..+7
    af.q[1] = *(const uint4*)&As[cur][m * 32 + 16 + 8 * hi];  // k = 16+8*hi..+7
#pragma unroll
    for (int t = 0; t < 8; ++t) {
      Frag bf;
      int n = t * 16 + ln;
      bf.q[0] = *(const uint4*)&Bs[cur][n * 32 + 8 * hi];     // k = 8*hi..+7
      bf.q[1] = *(const uint4*)&Bs[cur][n * 32 + 16 + 8 * hi];
      acc[t] = __builtin_amdgcn_wmma_f32_16x16x32_bf16(
          false, af.v, false, bf.v, (short)0, acc[t], false, false);
    }
  };

  int ntiles = K >> 5;
  stage(0, 0);
  // branch-free steady state: prefetch next, wait current, compute
  for (int i = 0; i < ntiles - 1; ++i) {
    stage((i + 1) * 32, (i + 1) & 1);
    wait_async4();                  // current tile pair complete, prefetch in flight
    __syncthreads();
    compute(i & 1);
    __syncthreads();                // all reads done before buffer reuse
  }
  wait_async0();
  __syncthreads();
  compute((ntiles - 1) & 1);

#pragma unroll
  for (int t = 0; t < 8; ++t)
#pragma unroll
    for (int r = 0; r < 8; ++r)
      C[(size_t)(m0 + wid * 16 + r + 8 * hi) * N + n0 + t * 16 + ln] = acc[t][r];
}

// ---------------- flash attention (GQA, causal) ----------------
// Q: (SEQ, NH*HD) bf16 rope'd ; Kc: (SEQ, NKV*HD) bf16 ; Vt: (NKV*HD, SEQ) bf16
// O: (SEQ, NH*HD) bf16
__global__ __launch_bounds__(128) void flash_attn_kernel(
    const unsigned short* __restrict__ Q, const unsigned short* __restrict__ Kc,
    const unsigned short* __restrict__ Vt, unsigned short* __restrict__ O) {
  __shared__ alignas(16) unsigned short Ks[2][32 * 128];   // [key][d]
  __shared__ alignas(16) unsigned short Vs[2][128 * 32];   // [d][key]
  __shared__ float Sc[4][16][33];                          // per-wave scores
  __shared__ alignas(16) unsigned short Ps[4][16][32];     // per-wave P (bf16)
  __shared__ float rowM[4][16], rowL[4][16], rowC[4][16];

  int tid = threadIdx.x;
  int wid = tid >> 5;
  int lane = tid & 31, ln = lane & 15, hi = lane >> 4;
  int head = blockIdx.y, kvh = head >> 2;               // N_REP = 4
  int qr = blockIdx.x * 64 + wid * 16;
  const float scale = 0.08838834764831845f;             // 1/sqrt(128)

  // preload Q fragments (4 k-chunks of 32 over head_dim 128)
  Frag qf[4];
#pragma unroll
  for (int c = 0; c < 4; ++c) {
    size_t base = (size_t)(qr + ln) * DIM + head * HD + c * 32;
    qf[c].q[0] = *(const uint4*)&Q[base + 8 * hi];
    qf[c].q[1] = *(const uint4*)&Q[base + 16 + 8 * hi];
  }

  v8f o[8];
#pragma unroll
  for (int t = 0; t < 8; ++t)
#pragma unroll
    for (int r = 0; r < 8; ++r) o[t][r] = 0.0f;

  if (lane < 16) {
    rowM[wid][ln] = -3.0e38f;
    rowL[wid][ln] = 0.0f;
  }

  // per-thread share of one KV block = 8 async b128 copies
  auto stage = [&](int kb, int buf) {
#pragma unroll
    for (int i = 0; i < 4; ++i) {           // K block: 32 keys x 128 d
      int idx = tid + 128 * i;              // 0..511
      int r = idx >> 4, c4 = idx & 15;
      async_copy_b128(lds_off(&Ks[buf][r * 128 + c4 * 8]),
                      &Kc[(size_t)(kb * 32 + r) * (NKV * HD) + kvh * HD + c4 * 8]);
    }
#pragma unroll
    for (int i = 0; i < 4; ++i) {           // V block from Vt: 128 d x 32 keys
      int idx = tid + 128 * i;
      int r = idx >> 2, c4 = idx & 3;
      async_copy_b128(lds_off(&Vs[buf][r * 32 + c4 * 8]),
                      &Vt[(size_t)(kvh * HD + r) * SEQ + kb * 32 + c4 * 8]);
    }
  };

  auto compute = [&](int kb, int buf) {
    // S = Q * K^T  (two 16-key column tiles)
#pragma unroll
    for (int t = 0; t < 2; ++t) {
      v8f s;
#pragma unroll
      for (int r = 0; r < 8; ++r) s[r] = 0.0f;
#pragma unroll
      for (int c = 0; c < 4; ++c) {
        Frag kf;                                        // B frag: lane=key, e over d
        int krow = t * 16 + ln;
        kf.q[0] = *(const uint4*)&Ks[buf][krow * 128 + c * 32 + 8 * hi];
        kf.q[1] = *(const uint4*)&Ks[buf][krow * 128 + c * 32 + 16 + 8 * hi];
        s = __builtin_amdgcn_wmma_f32_16x16x32_bf16(
            false, qf[c].v, false, kf.v, (short)0, s, false, false);
      }
      int key = kb * 32 + t * 16 + ln;
#pragma unroll
      for (int r = 0; r < 8; ++r) {
        int qrow = qr + r + 8 * hi;
        float v = s[r] * scale;
        if (key > qrow) v = -__builtin_inff();          // causal mask
        Sc[wid][r + 8 * hi][t * 16 + ln] = v;
      }
    }
    __syncthreads();

    // online softmax, one row per lane (lanes 0..15)
    if (lane < 16) {
      float mo = rowM[wid][ln], lo = rowL[wid][ln];
      float mx = mo;
#pragma unroll
      for (int c2 = 0; c2 < 32; ++c2) mx = fmaxf(mx, Sc[wid][ln][c2]);
      float corr = __expf(mo - mx);
      float sum = 0.0f;
#pragma unroll
      for (int c2 = 0; c2 < 32; ++c2) {
        float p = __expf(Sc[wid][ln][c2] - mx);
        sum += p;
        Ps[wid][ln][c2] = f2bf(p);
      }
      rowM[wid][ln] = mx;
      rowL[wid][ln] = lo * corr + sum;
      rowC[wid][ln] = corr;
    }
    __syncthreads();

    // rescale running O by per-row correction
#pragma unroll
    for (int r = 0; r < 8; ++r) {
      float cr = rowC[wid][r + 8 * hi];
#pragma unroll
      for (int t = 0; t < 8; ++t) o[t][r] *= cr;
    }

    // P fragment (A layout) from LDS
    Frag pf;
    pf.q[0] = *(const uint4*)&Ps[wid][ln][8 * hi];
    pf.q[1] = *(const uint4*)&Ps[wid][ln][16 + 8 * hi];

    // O += P * V   (8 output d-tiles of 16)
#pragma unroll
    for (int t = 0; t < 8; ++t) {
      Frag vf;                                          // B frag: lane=d, e over key
      int vrow = t * 16 + ln;
      vf.q[0] = *(const uint4*)&Vs[buf][vrow * 32 + 8 * hi];
      vf.q[1] = *(const uint4*)&Vs[buf][vrow * 32 + 16 + 8 * hi];
      o[t] = __builtin_amdgcn_wmma_f32_16x16x32_bf16(
          false, pf.v, false, vf.v, (short)0, o[t], false, false);
    }
  };

  int kbmax = blockIdx.x * 2 + 1;                       // causal upper block
  stage(0, 0);
  for (int kb = 0; kb < kbmax; ++kb) {
    stage(kb + 1, (kb + 1) & 1);    // prefetch next KV block
    wait_async8();                  // current block resident, prefetch in flight
    __syncthreads();
    compute(kb, kb & 1);
    __syncthreads();                // reads done before buffer reuse
  }
  wait_async0();
  __syncthreads();
  compute(kbmax, kbmax & 1);
  __syncthreads();

  // finalize: divide by row sum, emit bf16
#pragma unroll
  for (int r = 0; r < 8; ++r) {
    float inv = 1.0f / rowL[wid][r + 8 * hi];
#pragma unroll
    for (int t = 0; t < 8; ++t)
      O[(size_t)(qr + r + 8 * hi) * DIM + head * HD + t * 16 + ln] =
          f2bf(o[t][r] * inv);
  }
}

// ---------------- host-side launch ----------------
extern "C" void kernel_launch(void* const* d_in, const int* in_sizes, int n_in,
                              void* d_out, int out_size, void* d_ws, size_t ws_size,
                              hipStream_t stream) {
  const float* x  = (const float*)d_in[0];
  const float* wq = (const float*)d_in[1];
  const float* wk = (const float*)d_in[2];
  const float* wv = (const float*)d_in[3];
  const float* wo = (const float*)d_in[4];
  const float* fc = (const float*)d_in[5];
  const float* fs = (const float*)d_in[6];
  (void)in_sizes; (void)n_in; (void)out_size; (void)ws_size;

  char* ws = (char*)d_ws;
  size_t off = 0;
  auto alloc = [&](size_t bytes) -> void* {
    void* p = ws + off;
    off += (bytes + 255) & ~(size_t)255;
    return p;
  };

  unsigned short* xb   = (unsigned short*)alloc((size_t)SEQ * DIM * 2);
  unsigned short* wqt  = (unsigned short*)alloc((size_t)DIM * DIM * 2);        // [N][K]
  unsigned short* wkt  = (unsigned short*)alloc((size_t)(NKV * HD) * DIM * 2); // [N][K]
  unsigned short* wvt  = (unsigned short*)alloc((size_t)(NKV * HD) * DIM * 2); // [N][K]
  unsigned short* wot  = (unsigned short*)alloc((size_t)DIM * DIM * 2);        // [N][K]
  float*          qf32 = (float*)alloc((size_t)SEQ * DIM * 4);
  float*          kf32 = (float*)alloc((size_t)SEQ * (NKV * HD) * 4);
  float*          vf32 = (float*)alloc((size_t)SEQ * (NKV * HD) * 4);
  unsigned short* qbf  = (unsigned short*)alloc((size_t)SEQ * DIM * 2);
  unsigned short* kbf  = (unsigned short*)alloc((size_t)SEQ * (NKV * HD) * 2);
  unsigned short* vt   = (unsigned short*)alloc((size_t)(NKV * HD) * SEQ * 2); // [d][seq]
  unsigned short* aob  = (unsigned short*)alloc((size_t)SEQ * DIM * 2);

  // x -> bf16 (row-major, GEMM A operand)
  cvt_bf16_kernel<<<8192, 256, 0, stream>>>(x, xb, SEQ * DIM);

  // weights -> bf16 transposed [N][K] for TN GEMM
  transpose_cvt_kernel<<<dim3(DIM / 32, DIM / 32), 256, 0, stream>>>(wq, wqt, DIM, DIM);
  transpose_cvt_kernel<<<dim3((NKV * HD) / 32, DIM / 32), 256, 0, stream>>>(wk, wkt, DIM, NKV * HD);
  transpose_cvt_kernel<<<dim3((NKV * HD) / 32, DIM / 32), 256, 0, stream>>>(wv, wvt, DIM, NKV * HD);
  transpose_cvt_kernel<<<dim3(DIM / 32, DIM / 32), 256, 0, stream>>>(wo, wot, DIM, DIM);

  // projections
  gemm_bf16_tn_kernel<<<dim3(DIM / 128, SEQ / 128), 256, 0, stream>>>(
      xb, wqt, qf32, SEQ, DIM, DIM);
  gemm_bf16_tn_kernel<<<dim3((NKV * HD) / 128, SEQ / 128), 256, 0, stream>>>(
      xb, wkt, kf32, SEQ, NKV * HD, DIM);
  gemm_bf16_tn_kernel<<<dim3((NKV * HD) / 128, SEQ / 128), 256, 0, stream>>>(
      xb, wvt, vf32, SEQ, NKV * HD, DIM);

  // RoPE + bf16 for Q,K ; V -> transposed bf16 [d][seq]
  int tq = SEQ * NH * (HD / 2);
  rope_kernel<<<(tq + 255) / 256, 256, 0, stream>>>(qf32, qbf, fc, fs, NH, tq);
  int tk = SEQ * NKV * (HD / 2);
  rope_kernel<<<(tk + 255) / 256, 256, 0, stream>>>(kf32, kbf, fc, fs, NKV, tk);
  transpose_cvt_kernel<<<dim3((NKV * HD) / 32, SEQ / 32), 256, 0, stream>>>(
      vf32, vt, SEQ, NKV * HD);

  // flash attention: 32 q-blocks of 64 rows x 32 heads
  flash_attn_kernel<<<dim3(SEQ / 64, NH), 128, 0, stream>>>(qbf, kbf, vt, aob);

  // output projection -> f32 d_out
  gemm_bf16_tn_kernel<<<dim3(DIM / 128, SEQ / 128), 256, 0, stream>>>(
      aob, wot, (float*)d_out, SEQ, DIM, DIM);
}
